// BiLSTM_78958678769799
// MI455X (gfx1250) — matrix operand order repaired and checked by reading, not compiled, not run on hardware
//
#include <hip/hip_runtime.h>
#include <stdint.h>

#define Bsz 64
#define Ssz 512
#define Fsz 512
#define Hsz 512
#define Gsz 2048            // 4*H
#define AROWSTR (Fsz * 2 + 16)   // 1040 B: padded LDS row stride (bank-conflict free)

typedef __attribute__((ext_vector_type(16))) __bf16 v16bf;
typedef __attribute__((ext_vector_type(8)))  float  v8f;
typedef __attribute__((ext_vector_type(4)))  unsigned int v4u;
typedef __attribute__((ext_vector_type(8)))  int    v8i;
typedef __attribute__((ext_vector_type(4)))  int    v4i;

union BF16Frag { v16bf v; unsigned int u[8]; };

__device__ __forceinline__ uint16_t f2bf(float f) {
  unsigned int u = __float_as_uint(f);
  u += 0x7FFFu + ((u >> 16) & 1u);   // round-to-nearest-even (inputs finite)
  return (uint16_t)(u >> 16);
}

__device__ __forceinline__ float sigmoidf_(float x) {
  return 1.0f / (1.0f + __expf(-x));
}

// ---------------------------------------------------------------------------
// TDM: DMA a row-contiguous 16-row x 1024B tile (16 KB) from global into LDS,
// inserting 16B of padding every 1024B so LDS rows are AROWSTR apart.
// D# per CDNA5 ISA: data_size=8B, tensor_dim0=tile_dim0=2048,
// pad_interval=7 (256 DW = 1024B), pad_amount=3 (4 DW = 16B).
// Must be called by exactly one wave per block; follow with __syncthreads().
__device__ __forceinline__ void tdm_load_16rows(const void* gsrc, uint32_t lds_off) {
#if defined(__HIP_DEVICE_COMPILE__)
  uint64_t ga = (uint64_t)gsrc;
  v4u g0;
  g0[0] = 1u;                                                  // count=1 (valid D#)
  g0[1] = lds_off;                                             // lds_addr
  g0[2] = (uint32_t)ga;                                        // global_addr[31:0]
  g0[3] = (uint32_t)((ga >> 32) & 0x01FFFFFFu) | (2u << 30);   // addr[56:32], type=2
  v8i g1;
  g1[0] = (int)((3u << 16) | (1u << 20) | (7u << 22) | (3u << 25)); // dsize=8B,pad
  g1[1] = (int)(2048u << 16);   // tensor_dim0 = 2048 (8B units)
  g1[2] = (int)(1u << 16);      // tensor_dim1 = 1
  g1[3] = (int)(2048u << 16);   // tile_dim0 = 2048
  g1[4] = 1;                    // tile_dim1 = 1
  g1[5] = 2048;                 // tensor_dim0_stride
  g1[6] = 0;
  g1[7] = 0;
  v4i z4 = {0, 0, 0, 0};
#if __clang_major__ >= 23
  v8i z8 = {0, 0, 0, 0, 0, 0, 0, 0};
  __builtin_amdgcn_tensor_load_to_lds(g0, g1, z4, z4, z8, 0);
#else
  __builtin_amdgcn_tensor_load_to_lds(g0, g1, z4, z4, 0);
#endif
  __builtin_amdgcn_s_wait_tensorcnt(0);
#else
  (void)gsrc; (void)lds_off;
#endif
}

// Build an A-operand fragment (16x32 bf16, contiguous k = 16*half + s) from the padded
// LDS tile: two 16B ds loads per lane.
__device__ __forceinline__ BF16Frag lds_a_frag(const uint8_t* smA, int l16,
                                               int half, int kb) {
  BF16Frag a;
  const uint8_t* row = smA + l16 * AROWSTR;
  uint4 a0 = *(const uint4*)(row + (kb + half * 16) * 2);
  uint4 a1 = *(const uint4*)(row + (kb + half * 16 + 8) * 2);
  a.u[0] = a0.x; a.u[1] = a0.y; a.u[2] = a0.z; a.u[3] = a0.w;
  a.u[4] = a1.x; a.u[5] = a1.y; a.u[6] = a1.z; a.u[7] = a1.w;
  return a;
}

// ---------------------------------------------------------------------------
__global__ void bilstm_init_state(uint16_t* __restrict__ hbuf,
                                  float* __restrict__ cbuf) {
  int idx = blockIdx.x * blockDim.x + threadIdx.x;
  if (idx < 2 * 2 * Bsz * Hsz) hbuf[idx] = 0;
  if (idx < 2 * Bsz * Hsz)     cbuf[idx] = 0.0f;
}

// x (B,S,F) f32 -> xs (S,B,F) bf16
__global__ void bilstm_cast_x(const float* __restrict__ x,
                              uint16_t* __restrict__ xs) {
  long idx = (long)blockIdx.x * blockDim.x + threadIdx.x;
  int  f   = (int)(idx % Fsz);
  long rem = idx / Fsz;
  int  t   = (int)(rem % Ssz);
  int  b   = (int)(rem / Ssz);
  xs[((long)t * Bsz + b) * Fsz + f] = f2bf(x[idx]);
}

// W/U (K,4H) f32 -> transposed (4H,K) bf16
__global__ void bilstm_cast_w(const float* __restrict__ Wf,
                              const float* __restrict__ Uf,
                              const float* __restrict__ Wb,
                              const float* __restrict__ Ub,
                              uint16_t* __restrict__ wt,
                              uint16_t* __restrict__ ut) {
  long idx = (long)blockIdx.x * blockDim.x + threadIdx.x;
  int  sel = (int)(idx / ((long)Fsz * Gsz));
  long rem = idx % ((long)Fsz * Gsz);
  int  fr  = (int)(rem / Gsz);
  int  g   = (int)(rem % Gsz);
  const float* src = (sel == 0) ? Wf : (sel == 1) ? Wb : (sel == 2) ? Uf : Ub;
  uint16_t* dst = (sel < 2) ? (wt + (long)sel * Gsz * Fsz)
                            : (ut + (long)(sel - 2) * Gsz * Hsz);
  dst[(long)g * Fsz + fr] = f2bf(src[(long)fr * Gsz + g]);
}

// ---------------------------------------------------------------------------
// Pre-GEMM: pre[dir][t][b][g] = xs[t][b][:] @ W_dir[:,g] + bias_dir[g]
// (32768 x 2048, K=512). Block = 8 waves sharing one 16-row A tile (TDM->LDS);
// each wave does a 16x64 strip with double-buffered B fragments.
__global__ void __launch_bounds__(256) bilstm_pregemm(
    const uint16_t* __restrict__ xs, const uint16_t* __restrict__ wt,
    const float* __restrict__ bias_f, const float* __restrict__ bias_b,
    float* __restrict__ pre) {
  __shared__ __align__(16) uint8_t smA[16 * AROWSTR];

  const int wv   = threadIdx.x >> 5;
  const int lane = threadIdx.x & 31;
  const int half = lane >> 4;
  const int l16  = lane & 15;
  const int blk  = blockIdx.x;
  const int dir  = blk >> 13;            // 8192 blocks per direction
  const int rem  = blk & 8191;
  const int mtile = rem >> 2;            // 0..2047 : 16-row tile of (S*B)
  const int n0    = ((rem & 3) * 8 + wv) * 64;

  // Stage A tile (shared by both dirs, all 8 waves) via Tensor Data Mover.
  if (wv == 0)
    tdm_load_16rows(xs + (size_t)mtile * 16 * Fsz, (uint32_t)(uintptr_t)smA);
  __syncthreads();

  const uint16_t* W    = wt + (long)dir * Fsz * Gsz;   // (G,F) bf16
  const float*    bias = dir ? bias_b : bias_f;
  float*          out  = pre + (long)dir * Ssz * Bsz * Gsz;

  v8f acc[4];
#pragma unroll
  for (int t4 = 0; t4 < 4; ++t4) {
    float bv = bias[n0 + t4 * 16 + l16];
#pragma unroll
    for (int e = 0; e < 8; ++e) acc[t4][e] = bv;
  }

  auto loadB = [&](int kb, BF16Frag* dst) {
#pragma unroll
    for (int t4 = 0; t4 < 4; ++t4) {
      const long col = n0 + t4 * 16 + l16;
#pragma unroll
      for (int v = 0; v < 8; ++v) {      // B 32x16: lanes 0-15 K0-15, 16-31 K16-31
        int kk = kb + half * 16 + v * 2;
        dst[t4].u[v] = *(const unsigned int*)(W + col * Fsz + kk);
      }
    }
  };

  BF16Frag bm[2][4];
  loadB(0, bm[0]);
  for (int kb = 0; kb < Fsz; kb += 32) {
    const int cur = (kb >> 5) & 1;
    if (kb + 32 < Fsz) loadB(kb + 32, bm[cur ^ 1]);   // prefetch next K-slab
    BF16Frag a = lds_a_frag(smA, l16, half, kb);
#pragma unroll
    for (int t4 = 0; t4 < 4; ++t4)
      acc[t4] = __builtin_amdgcn_wmma_f32_16x16x32_bf16(
          false, a.v, false, bm[cur][t4].v, (short)0, acc[t4], false, false);
  }

#pragma unroll
  for (int t4 = 0; t4 < 4; ++t4) {
    const long col = n0 + t4 * 16 + l16;
#pragma unroll
    for (int e = 0; e < 8; ++e) {                     // C/D: row = 8*half + vgpr
      long mr = (long)mtile * 16 + half * 8 + e;
      out[mr * Gsz + col] = acc[t4][e];
    }
  }
}

// ---------------------------------------------------------------------------
// One recurrence step, both directions. gates = pre[t] + h_prev @ U; wave owns
// the same 16 hidden cols of all four gate blocks -> fused local cell update.
__global__ void __launch_bounds__(256) bilstm_step(
    const float* __restrict__ pre, const uint16_t* __restrict__ ut,
    uint16_t* __restrict__ hbuf, float* __restrict__ cbuf,
    float* __restrict__ out, int t) {
  __shared__ __align__(16) uint8_t smA[16 * AROWSTR];

  const int wv   = threadIdx.x >> 5;
  const int lane = threadIdx.x & 31;
  const int half = lane >> 4;
  const int l16  = lane & 15;
  const int blk  = blockIdx.x;
  const int dir  = blk >> 4;             // 16 blocks per direction
  const int rem  = blk & 15;
  const int mtile = rem >> 2;            // batch tile 0..3 (block-uniform)
  const int j0    = ((rem & 3) * 8 + wv) * 16;

  const int par = t & 1;                 // h double-buffered across steps
  const uint16_t* hprev = hbuf + (long)(dir * 2 + par) * Bsz * Hsz;
  uint16_t*       hnext = hbuf + (long)(dir * 2 + (par ^ 1)) * Bsz * Hsz;
  float* cst = cbuf + (long)dir * Bsz * Hsz;

  const int tx = dir ? (Ssz - 1 - t) : t;
  const float* prew = pre + ((long)dir * Ssz * Bsz + (long)tx * Bsz) * Gsz;
  const uint16_t* U = ut + (long)dir * Gsz * Hsz;     // (G,H) bf16

  // Stage h_prev rows for this batch tile via TDM (shared by 8 waves).
  if (wv == 0)
    tdm_load_16rows(hprev + (size_t)mtile * 16 * Hsz, (uint32_t)(uintptr_t)smA);
  __syncthreads();

  v8f acc[4];
#pragma unroll
  for (int gi = 0; gi < 4; ++gi) {       // init accum with x@W + b tile
    const long col = (long)gi * Hsz + j0 + l16;
#pragma unroll
    for (int e = 0; e < 8; ++e)
      acc[gi][e] = prew[((long)mtile * 16 + half * 8 + e) * Gsz + col];
  }

  auto loadB = [&](int kb, BF16Frag* dst) {
#pragma unroll
    for (int gi = 0; gi < 4; ++gi) {
      const long col = (long)gi * Hsz + j0 + l16;
#pragma unroll
      for (int v = 0; v < 8; ++v) {
        int kk = kb + half * 16 + v * 2;
        dst[gi].u[v] = *(const unsigned int*)(U + col * Hsz + kk);
      }
    }
  };

  BF16Frag bm[2][4];
  loadB(0, bm[0]);
  for (int kb = 0; kb < Hsz; kb += 32) {
    const int cur = (kb >> 5) & 1;
    if (kb + 32 < Hsz) loadB(kb + 32, bm[cur ^ 1]);
    BF16Frag a = lds_a_frag(smA, l16, half, kb);
#pragma unroll
    for (int gi = 0; gi < 4; ++gi)
      acc[gi] = __builtin_amdgcn_wmma_f32_16x16x32_bf16(
          false, a.v, false, bm[cur][gi].v, (short)0, acc[gi], false, false);
  }

#pragma unroll
  for (int e = 0; e < 8; ++e) {
    const int brow = mtile * 16 + half * 8 + e;
    const int j    = j0 + l16;
    float iv = sigmoidf_(acc[0][e]);
    float fv = sigmoidf_(acc[1][e]);
    float gv = tanhf(acc[2][e]);
    float ov = sigmoidf_(acc[3][e]);
    float cold = cst[(long)brow * Hsz + j];
    float cnew = fv * cold + iv * gv;
    cst[(long)brow * Hsz + j] = cnew;
    float hv = ov * tanhf(cnew);
    // hidden_seq (B,S,2H): backward half stored un-reversed (faithful bug)
    out[(long)brow * (Ssz * 2 * Hsz) + (long)t * (2 * Hsz) + dir * Hsz + j] = hv;
    hnext[(long)brow * Hsz + j] = f2bf(hv);
    if (t == Ssz - 1) {
      const long hb = (long)Bsz * Ssz * 2 * Hsz;
      out[hb + (long)brow * (2 * Hsz) + dir * Hsz + j] = hv;      // h_t
      if (dir == 0) {                                             // c_t = [cTf,cTf]
        const long cb = hb + (long)Bsz * 2 * Hsz;
        out[cb + (long)brow * (2 * Hsz) + j] = cnew;
        out[cb + (long)brow * (2 * Hsz) + Hsz + j] = cnew;
      }
    }
  }
}

// ---------------------------------------------------------------------------
extern "C" void kernel_launch(void* const* d_in, const int* in_sizes, int n_in,
                              void* d_out, int out_size, void* d_ws, size_t ws_size,
                              hipStream_t stream) {
  const float* x  = (const float*)d_in[0];
  const float* Wf = (const float*)d_in[1];
  const float* Uf = (const float*)d_in[2];
  const float* bf = (const float*)d_in[3];
  const float* Wb = (const float*)d_in[4];
  const float* Ub = (const float*)d_in[5];
  const float* bb = (const float*)d_in[6];
  float* out = (float*)d_out;

  char* ws = (char*)d_ws;
  size_t off = 0;
  uint16_t* xs   = (uint16_t*)(ws + off); off += (size_t)Ssz * Bsz * Fsz * 2;
  uint16_t* wt   = (uint16_t*)(ws + off); off += (size_t)2 * Fsz * Gsz * 2;
  uint16_t* ut   = (uint16_t*)(ws + off); off += (size_t)2 * Hsz * Gsz * 2;
  float*    pre  = (float*)(ws + off);    off += (size_t)2 * Ssz * Bsz * Gsz * 4;
  uint16_t* hbuf = (uint16_t*)(ws + off); off += (size_t)2 * 2 * Bsz * Hsz * 2;
  float*    cbuf = (float*)(ws + off);    off += (size_t)2 * Bsz * Hsz * 4;
  (void)ws_size; (void)in_sizes; (void)n_in; (void)out_size;

  bilstm_init_state<<<512, 256, 0, stream>>>(hbuf, cbuf);
  bilstm_cast_x<<<(Bsz * Ssz * Fsz) / 256, 256, 0, stream>>>(x, xs);
  bilstm_cast_w<<<(4 * Fsz * Gsz) / 256, 256, 0, stream>>>(Wf, Uf, Wb, Ub, wt, ut);
  bilstm_pregemm<<<16384, 256, 0, stream>>>(xs, wt, bf, bb, pre);
  for (int t = 0; t < Ssz; ++t)
    bilstm_step<<<32, 256, 0, stream>>>(pre, ut, hbuf, cbuf, out, t);
}
